// Head_73607149519532
// MI455X (gfx1250) — matrix operand order repaired and checked by reading, not compiled, not run on hardware
//
#include <hip/hip_runtime.h>

typedef _Float16 h16;
typedef __attribute__((ext_vector_type(16))) _Float16 v16h;
typedef __attribute__((ext_vector_type(8)))  float    v8f;
typedef __attribute__((ext_vector_type(4)))  unsigned int u32x4;
typedef __attribute__((ext_vector_type(8)))  int      i32x8;
typedef __attribute__((ext_vector_type(4)))  int      i32x4;

#define HEAD_SIZE 64
#define EMBED_SIZE 512
#define VOCAB_SIZE 32000
#define BB 4
#define TT 2048
#define BT (BB * TT)   // 8192

// ---- CDNA5 TDM availability ------------------------------------------------
#if defined(__has_builtin)
#if __has_builtin(__builtin_amdgcn_tensor_load_to_lds) && \
    __has_builtin(__builtin_amdgcn_s_wait_tensorcnt)
#define HAVE_TDM 1
#endif
#endif
#ifndef HAVE_TDM
#define HAVE_TDM 0
#endif

#if HAVE_TDM
// 1-D TDM copy: nelem (=2048) f16 elements, global -> LDS. D# per ISA ch.8.
__device__ __forceinline__ void tdm_load_1d(unsigned int lds_off,
                                            const void* gptr) {
  unsigned long long ga = (unsigned long long)(uintptr_t)gptr;
  u32x4 g0;
  g0[0] = 1u;                                   // count=1, user descriptor
  g0[1] = lds_off;                              // lds_addr (bytes)
  g0[2] = (unsigned int)(ga & 0xFFFFFFFFu);     // global_addr[31:0]
  g0[3] = (unsigned int)((ga >> 32) & 0x1FFFFFFu) | (2u << 30);  // type=2
  i32x8 g1;
  g1[0] = (int)(1u << 16);          // data_size=1 (2 bytes), wg_mask=0
  g1[1] = (int)(2048u << 16);       // tensor_dim0[15:0]
  g1[2] = (int)(1u << 16);          // tensor_dim0 hi=0, tensor_dim1 lo=1
  g1[3] = (int)(2048u << 16);       // tensor_dim1 hi=0, tile_dim0=2048
  g1[4] = 0;                        // tile_dim1=0, tile_dim2=0 (1-D)
  g1[5] = 2048;                     // tensor_dim0_stride
  g1[6] = 0;
  g1[7] = 0;
  i32x4 z = {0, 0, 0, 0};
#if __has_include(<hip/amd_detail/amd_gfx1250_TDM.h>)
  i32x8 z8 = {0, 0, 0, 0, 0, 0, 0, 0};          // therock toolchain: 6 args
  __builtin_amdgcn_tensor_load_to_lds(g0, g1, z, z, z8, 0);
#else
  __builtin_amdgcn_tensor_load_to_lds(g0, g1, z, z, 0);  // ROCm 7.2: 5 args
#endif
}
#endif

__device__ __forceinline__ v8f wmma_f16(v16h a, v16h b, v8f c) {
  return __builtin_amdgcn_wmma_f32_16x16x32_f16(
      false, a, false, b, (short)0, c, false, false);
}

// ---- fragment loaders (contiguous k = 16*(lane/16) + e, wave32) -------------------------

// A 16x32 f16 fragment from row-major [rows x ld] half source.
__device__ __forceinline__ v16h load_a_half(const h16* src, int row0, int ld,
                                            int k0, int lane) {
  v16h a;
  int m  = lane & 15;
  int kb = (lane < 16) ? 0 : 16;
  const h16* p = src + (size_t)(row0 + m) * ld + k0 + kb;
#pragma unroll
  for (int e = 0; e < 16; ++e) a[e] = p[e];
  return a;
}

// B 32x16 from row-major half [K x N] source.
__device__ __forceinline__ v16h load_b_rowmajor_h(const h16* src, int k0,
                                                  int n0, int ld, int lane) {
  v16h b;
  int n  = n0 + (lane & 15);
  int kb = k0 + ((lane >> 4) << 4);
  const h16* p = src + (size_t)kb * ld + n;
#pragma unroll
  for (int e = 0; e < 16; ++e) b[e] = p[(size_t)e * ld];
  return b;
}

// B 32x16 where B(k,n) = src[(n0+n)*ld + k0+k]  (K^T for Q.K^T); contiguous.
__device__ __forceinline__ v16h load_b_transposed_h(const h16* src, int n0,
                                                    int k0, int ld, int lane) {
  v16h b;
  int n  = n0 + (lane & 15);
  int kb = k0 + ((lane >> 4) << 4);
  const h16* p = src + (size_t)n * ld + kb;
#pragma unroll
  for (int e = 0; e < 16; ++e) b[e] = p[e];
  return b;
}

// ---- kernel 0: swizzle f32 [K x N] weights into f16 B-fragment stream ------
// fragment id = nt*(K/32) + kt ; per (fid,lane): 16 contiguous halves,
// element e -> (k = kt*32 + 16*(lane>>4) + e, n = nt*16 + lane&15)
__global__ void swizzle_b_kernel(const float* __restrict__ src,
                                 h16* __restrict__ dst, int K, int N) {
  int idx = blockIdx.x * blockDim.x + threadIdx.x;  // one per (fid, lane)
  int nfrag = (K / 32) * (N / 16);
  if (idx >= nfrag * 32) return;
  int fid = idx >> 5, lane = idx & 31;
  int kpf = K / 32;
  int nt = fid / kpf, kt = fid % kpf;
  int n  = nt * 16 + (lane & 15);
  int k0 = kt * 32 + ((lane >> 4) << 4);
  h16* out = dst + (size_t)idx * 16;
  const float* p = src + (size_t)k0 * N + n;
#pragma unroll
  for (int e = 0; e < 16; ++e) out[e] = (h16)p[(size_t)e * N];
}

// ---- kernel 1: embeddings + QKV projection (WMMA) --------------------------
// grid 512 x 128; block = 16 rows; wave w = H-col tile w; B from swizzled ws.
__global__ void qkv_kernel(const int* __restrict__ x,
                           const float* __restrict__ tok_emb,
                           const float* __restrict__ pos_emb,
                           const h16* __restrict__ wq_sw,
                           const h16* __restrict__ wk_sw,
                           const h16* __restrict__ wv_sw,
                           h16* __restrict__ qh, h16* __restrict__ kh,
                           h16* __restrict__ vh) {
  __shared__ h16 hlds[16 * EMBED_SIZE];  // 16 KB
  const int tid = threadIdx.x;
  const int rowbase = blockIdx.x * 16;

  for (int i = tid; i < 16 * EMBED_SIZE; i += 128) {
    int r = i >> 9, e = i & (EMBED_SIZE - 1);
    int row = rowbase + r;
    int t   = row & (TT - 1);
    int tok = x[row];
    hlds[i] = (h16)(tok_emb[(size_t)tok * EMBED_SIZE + e] +
                    pos_emb[(size_t)t * EMBED_SIZE + e]);
  }
  __syncthreads();

  const int wave = tid >> 5, lane = tid & 31;
  const int n0 = wave * 16;                    // nt = wave
  const int kpf = EMBED_SIZE / 32;             // 16
  v8f accq = {}, acck = {}, accv = {};

  for (int kt = 0; kt < kpf; ++kt) {
    v16h a;
    {
      int m  = lane & 15;
      int kb = kt * 32 + ((lane < 16) ? 0 : 16);
#pragma unroll
      for (int e = 0; e < 16; ++e)
        a[e] = hlds[m * EMBED_SIZE + kb + e];
    }
    size_t foff = (((size_t)wave * kpf + kt) * 32 + lane) * 16;
    v16h bq = *(const v16h*)(wq_sw + foff);
    v16h bk = *(const v16h*)(wk_sw + foff);
    v16h bv = *(const v16h*)(wv_sw + foff);
    accq = wmma_f16(a, bq, accq);
    acck = wmma_f16(a, bk, acck);
    accv = wmma_f16(a, bv, accv);
  }

#pragma unroll
  for (int r = 0; r < 8; ++r) {
    int grow = rowbase + r + ((lane >> 4) << 3);
    int col  = n0 + (lane & 15);
    qh[(size_t)grow * HEAD_SIZE + col] = (h16)accq[r];
    kh[(size_t)grow * HEAD_SIZE + col] = (h16)acck[r];
    vh[(size_t)grow * HEAD_SIZE + col] = (h16)accv[r];
  }
}

// ---- kernel 2: wave-level flash attention (causal, online softmax) ---------
__global__ void attn_kernel(const h16* __restrict__ qh,
                            const h16* __restrict__ kh,
                            const h16* __restrict__ vh,
                            h16* __restrict__ atth) {
  __shared__ h16 pss[16][32];
  const int lane = threadIdx.x;
  const int tile = blockIdx.x;
  const int b  = tile / (TT / 16);
  const int qi = tile % (TT / 16);
  const int qrow0 = b * TT + qi * 16;
  const int ncol  = lane & 15;
  const int rhalf = (lane >> 4) << 3;

  v16h aq0 = load_a_half(qh, qrow0, HEAD_SIZE, 0, lane);
  v16h aq1 = load_a_half(qh, qrow0, HEAD_SIZE, 32, lane);

  v8f accO[4];
  float mrun[8], lrun[8];
#pragma unroll
  for (int j = 0; j < 4; ++j) accO[j] = (v8f){};
#pragma unroll
  for (int r = 0; r < 8; ++r) { mrun[r] = -1e30f; lrun[r] = 0.f; }

  const int nk = (qi >> 1) + 1;
  for (int kb2 = 0; kb2 < nk; ++kb2) {
    const int kpos0 = kb2 * 32;
    const int krow0 = b * TT + kpos0;

    v8f s0 = {}, s1 = {};
    {
      v16h b0a = load_b_transposed_h(kh, krow0, 0, HEAD_SIZE, lane);
      v16h b0b = load_b_transposed_h(kh, krow0, 32, HEAD_SIZE, lane);
      s0 = wmma_f16(aq0, b0a, s0);
      s0 = wmma_f16(aq1, b0b, s0);
      v16h b1a = load_b_transposed_h(kh, krow0 + 16, 0, HEAD_SIZE, lane);
      v16h b1b = load_b_transposed_h(kh, krow0 + 16, 32, HEAD_SIZE, lane);
      s1 = wmma_f16(aq0, b1a, s1);
      s1 = wmma_f16(aq1, b1b, s1);
    }

    const float scale = 0.125f;
#pragma unroll
    for (int r = 0; r < 8; ++r) {
      int qp = qi * 16 + r + rhalf;
      float v0 = s0[r] * scale;
      float v1 = s1[r] * scale;
      if (kpos0 + ncol > qp)      v0 = -1e30f;
      if (kpos0 + 16 + ncol > qp) v1 = -1e30f;

      float tm = fmaxf(v0, v1);
#pragma unroll
      for (int off = 1; off < 16; off <<= 1)
        tm = fmaxf(tm, __shfl_xor(tm, off, 32));
      float nm   = fmaxf(mrun[r], tm);
      float corr = __expf(mrun[r] - nm);
      float p0 = __expf(v0 - nm), p1 = __expf(v1 - nm);
      float rs = p0 + p1;
#pragma unroll
      for (int off = 1; off < 16; off <<= 1)
        rs += __shfl_xor(rs, off, 32);
      lrun[r] = lrun[r] * corr + rs;
      mrun[r] = nm;
      accO[0][r] *= corr; accO[1][r] *= corr;
      accO[2][r] *= corr; accO[3][r] *= corr;
      s0[r] = p0; s1[r] = p1;
    }

    __syncthreads();
#pragma unroll
    for (int r = 0; r < 8; ++r) {
      int row = r + rhalf;
      pss[row][ncol]      = (h16)s0[r];
      pss[row][ncol + 16] = (h16)s1[r];
    }
    __syncthreads();
    v16h pa;
    {
      int m  = lane & 15;
      int kb = (lane < 16) ? 0 : 16;
#pragma unroll
      for (int e = 0; e < 16; ++e)
        pa[e] = pss[m][kb + e];
    }
#pragma unroll
    for (int j = 0; j < 4; ++j) {
      v16h bv = load_b_rowmajor_h(vh, krow0, j * 16, HEAD_SIZE, lane);
      accO[j] = wmma_f16(pa, bv, accO[j]);
    }
  }

#pragma unroll
  for (int r = 0; r < 8; ++r) {
    float inv = 1.0f / lrun[r];
    int grow = qrow0 + r + rhalf;
#pragma unroll
    for (int j = 0; j < 4; ++j)
      atth[(size_t)grow * HEAD_SIZE + j * 16 + ncol] =
          (h16)(accO[j][r] * inv);
  }
}

// ---- kernel 3: logits GEMM + fused online logsumexp ------------------------
// grid 128 x 256; block = 64 rows x full V. 8 waves = 4 row-tiles x 2 col-
// halves. B chunks (4KB = 2 col-tiles x K=64) TDM'd into LDS, double-buffered.
#define LNITER (VOCAB_SIZE / 32)  // 1000
__global__ void logits_kernel(const h16* __restrict__ atth,
                              const h16* __restrict__ wlm_sw,
                              const float* __restrict__ b_lm,
                              float* __restrict__ logits,
                              float* __restrict__ lse) {
  __shared__ __align__(32) h16 bufB[2][2048];       // 2 x 4KB B chunks
  __shared__ __align__(16) float rowbuf[4][16][32]; // 8KB store staging
  __shared__ float smm[8][16], sml[8][16];
  const int tid  = threadIdx.x;
  const int wave = tid >> 5, lane = tid & 31;
  const int rt = wave & 3, ch = wave >> 2;
  const int rowbase = blockIdx.x * 64;
  const int row0 = rowbase + rt * 16;
  const int ncol  = lane & 15;
  const int rhalf = (lane >> 4) << 3;

  v16h a0 = load_a_half(atth, row0, HEAD_SIZE, 0, lane);
  v16h a1 = load_a_half(atth, row0, HEAD_SIZE, 32, lane);

  float mrun[8], lrun[8];
#pragma unroll
  for (int r = 0; r < 8; ++r) { mrun[r] = -1e30f; lrun[r] = 0.f; }

#if HAVE_TDM
  if (wave == 0)
    tdm_load_1d((unsigned int)(uintptr_t)&bufB[0][0], wlm_sw);
#endif

  for (int i = 0; i < LNITER; ++i) {
#if HAVE_TDM
    const int db = i & 1;
    if (wave == 0) {
      if (i + 1 < LNITER) {
        tdm_load_1d((unsigned int)(uintptr_t)&bufB[db ^ 1][0],
                    wlm_sw + (size_t)(i + 1) * 2048);
        __builtin_amdgcn_s_wait_tensorcnt(1);
      } else {
        __builtin_amdgcn_s_wait_tensorcnt(0);
      }
    }
    __syncthreads();  // chunk i ready; rowbuf reads of i-1 done
    const h16* bf = &bufB[db][0];
#else
    __syncthreads();
    ((uint4*)&bufB[0][0])[tid] =
        ((const uint4*)(wlm_sw + (size_t)i * 2048))[tid];  // 256*16B = 4KB
    __syncthreads();
    const h16* bf = &bufB[0][0];
#endif
    // this wave's 2 K-fragments of its col tile (ch)
    v16h b0 = *(const v16h*)(bf + ((ch * 2 + 0) * 32 + lane) * 16);
    v16h b1 = *(const v16h*)(bf + ((ch * 2 + 1) * 32 + lane) * 16);
    v8f acc = {};
    acc = wmma_f16(a0, b0, acc);
    acc = wmma_f16(a1, b1, acc);

    const int n0 = i * 32 + ch * 16;
    const float bias = b_lm[n0 + ncol];
#pragma unroll
    for (int r = 0; r < 8; ++r) {
      float vl = acc[r] + bias;
      float nm = fmaxf(mrun[r], vl);
      lrun[r] = lrun[r] * __expf(mrun[r] - nm) + __expf(vl - nm);
      mrun[r] = nm;
      rowbuf[rt][r + rhalf][ch * 16 + ncol] = vl;
    }
    __syncthreads();  // rowbuf tile (16 rows x 32 cols) complete
    // coalesced b128 stores: 16 rows x 128B per row-tile, split over 2 waves
#pragma unroll
    for (int j = 0; j < 2; ++j) {
      int g   = ch * 64 + j * 32 + lane;  // 0..127 segment id
      int rr  = g >> 3, seg = g & 7;
      float4 d = *(const float4*)&rowbuf[rt][rr][seg * 4];
      *(float4*)&logits[(size_t)(row0 + rr) * VOCAB_SIZE + i * 32 + seg * 4] = d;
    }
  }

  // per-row logsumexp: reduce across the 16 lanes of each half
#pragma unroll
  for (int r = 0; r < 8; ++r) {
#pragma unroll
    for (int off = 1; off < 16; off <<= 1) {
      float om = __shfl_xor(mrun[r], off, 32);
      float ol = __shfl_xor(lrun[r], off, 32);
      float nm = fmaxf(mrun[r], om);
      lrun[r] = lrun[r] * __expf(mrun[r] - nm) + ol * __expf(om - nm);
      mrun[r] = nm;
    }
  }
  if (ncol == 0) {
#pragma unroll
    for (int r = 0; r < 8; ++r) {
      smm[wave][rhalf + r] = mrun[r];
      sml[wave][rhalf + r] = lrun[r];
    }
  }
  __syncthreads();
  if (tid < 64) {  // combine the two col-half waves of each row-tile
    int rt2 = tid >> 4, r = tid & 15;
    float m = smm[rt2][r],     l = sml[rt2][r];
    float om = smm[4 + rt2][r], ol = sml[4 + rt2][r];
    float nm = fmaxf(m, om);
    l = l * __expf(m - nm) + ol * __expf(om - nm);
    lse[rowbase + tid] = nm + __logf(l);
  }
}

// ---- kernel 4: loss = mean(lse[row] - logit[row, target[row]]) -------------
__global__ void loss_kernel(const int* __restrict__ target,
                            const float* __restrict__ logits,
                            const float* __restrict__ lse,
                            float* __restrict__ loss_out) {
  __shared__ float red[1024];
  const int tid = threadIdx.x;
  float s = 0.f;
  for (int row = tid; row < BT; row += 1024) {
    int tgt = target[row];
    s += lse[row] - logits[(size_t)row * VOCAB_SIZE + tgt];
  }
  red[tid] = s;
  __syncthreads();
  for (int st = 512; st > 0; st >>= 1) {
    if (tid < st) red[tid] += red[tid + st];
    __syncthreads();
  }
  if (tid == 0) loss_out[0] = red[0] / (float)BT;
}

extern "C" void kernel_launch(void* const* d_in, const int* in_sizes, int n_in,
                              void* d_out, int out_size, void* d_ws,
                              size_t ws_size, hipStream_t stream) {
  (void)in_sizes; (void)n_in; (void)out_size; (void)ws_size;
  const int*   x       = (const int*)d_in[0];
  const int*   target  = (const int*)d_in[1];
  const float* tok_emb = (const float*)d_in[2];
  const float* pos_emb = (const float*)d_in[3];
  const float* Wq      = (const float*)d_in[4];
  const float* Wk      = (const float*)d_in[5];
  const float* Wv      = (const float*)d_in[6];
  const float* W_lm    = (const float*)d_in[7];
  const float* b_lm    = (const float*)d_in[8];

  float* logits = (float*)d_out;                       // [BT, V]
  float* loss   = logits + (size_t)BT * VOCAB_SIZE;    // scalar

  const size_t HSZ = (size_t)BT * HEAD_SIZE * sizeof(h16);      // 1 MB
  const size_t WSZ = (size_t)EMBED_SIZE * HEAD_SIZE * sizeof(h16);  // 64 KB
  char* w = (char*)d_ws;
  h16* qh    = (h16*)(w);
  h16* kh    = (h16*)(w + HSZ);
  h16* vh    = (h16*)(w + 2 * HSZ);
  h16* atth  = (h16*)(w + 3 * HSZ);
  float* lse = (float*)(w + 4 * HSZ);                  // [BT] f32 (32 KB)
  h16* wq_sw = (h16*)(w + 4 * HSZ + 32 * 1024);
  h16* wk_sw = (h16*)((char*)wq_sw + WSZ);
  h16* wv_sw = (h16*)((char*)wk_sw + WSZ);
  h16* wlm_sw = (h16*)((char*)wv_sw + WSZ);            // 4 MB

  // weight pre-swizzle (f32 row-major -> f16 fragment streams)
  {
    int thr = (EMBED_SIZE / 32) * (HEAD_SIZE / 16) * 32;       // 2048
    swizzle_b_kernel<<<(thr + 255) / 256, 256, 0, stream>>>(Wq, wq_sw,
                                                            EMBED_SIZE, HEAD_SIZE);
    swizzle_b_kernel<<<(thr + 255) / 256, 256, 0, stream>>>(Wk, wk_sw,
                                                            EMBED_SIZE, HEAD_SIZE);
    swizzle_b_kernel<<<(thr + 255) / 256, 256, 0, stream>>>(Wv, wv_sw,
                                                            EMBED_SIZE, HEAD_SIZE);
    int thr2 = (HEAD_SIZE / 32) * (VOCAB_SIZE / 16) * 32;      // 128000
    swizzle_b_kernel<<<(thr2 + 255) / 256, 256, 0, stream>>>(W_lm, wlm_sw,
                                                             HEAD_SIZE, VOCAB_SIZE);
  }

  qkv_kernel<<<BT / 16, 128, 0, stream>>>(x, tok_emb, pos_emb,
                                          wq_sw, wk_sw, wv_sw, qh, kh, vh);
  attn_kernel<<<BT / 16, 32, 0, stream>>>(qh, kh, vh, atth);
  logits_kernel<<<BT / 64, 256, 0, stream>>>(atth, wlm_sw, b_lm, logits, lse);
  loss_kernel<<<1, 1024, 0, stream>>>(target, logits, lse, loss);
}
